// Discriminator_26843545600235
// MI455X (gfx1250) — matrix operand order repaired and checked by reading, not compiled, not run on hardware
//
#include <hip/hip_runtime.h>
#include <hip/hip_bf16.h>

typedef __attribute__((ext_vector_type(16))) _Float16 v16h;
typedef __attribute__((ext_vector_type(8)))  _Float16 v8h;
typedef __attribute__((ext_vector_type(8)))  float    v8f;

__device__ __forceinline__ float leaky01(float v) { return v >= 0.f ? v : 0.01f * v; }

// ---------------- prep kernels ----------------

// conv1 weights: [16][1][6][6][6] f32 -> [16][224] f16 in the paired-kd enumeration:
// k -> (P=k/32, lh=(k%32)/16, j=k%16); J=P*16+j; tap kd=2*(J/36)+lh, kh=(J%36)/6, kw=J%6
__global__ void wpad1_kernel(const float* __restrict__ w, _Float16* __restrict__ wh) {
  int idx = blockIdx.x * blockDim.x + threadIdx.x;
  if (idx >= 16 * 224) return;
  int m = idx / 224, k = idx - m * 224;
  int P = k >> 5, q = k & 31, lhb = q >> 4, j = q & 15;
  int J = P * 16 + j;
  _Float16 v = (_Float16)0.f;
  if (J < 108) {
    int a = J / 36, rem = J % 36;
    int kd = 2 * a + lhb, kh = rem / 6, kw = rem % 6;
    v = (_Float16)w[m * 216 + kd * 36 + kh * 6 + kw];
  }
  wh[idx] = v;
}

// conv2-4 weights: [Cout][Cin][KV] f32 -> [Cout][Kpad] f16 with k=(t*Cin+c), zero padded
__global__ void wpad_cl_kernel(const float* __restrict__ w, _Float16* __restrict__ wh,
                               int Cout, int Cin, int KV, int Kpad) {
  int idx = blockIdx.x * blockDim.x + threadIdx.x;
  if (idx >= Cout * Kpad) return;
  int m = idx / Kpad, k = idx - m * Kpad;
  _Float16 v = (_Float16)0.f;
  if (k < Cin * KV) {
    int t = k / Cin, c = k - t * Cin;
    v = (_Float16)w[(m * Cin + c) * KV + t];
  }
  wh[idx] = v;
}

// T [512][1024] f32 -> Tt [1024][512] f16 (transposed, so GEMM-B runs are contiguous)
__global__ void tpose_f16_kernel(const float* __restrict__ T, _Float16* __restrict__ Tt) {
  int idx = blockIdx.x * blockDim.x + threadIdx.x;
  if (idx >= 1024 * 512) return;
  int nn = idx >> 9, k = idx & 511;
  Tt[idx] = (_Float16)T[k * 1024 + nn];
}

// tap-offset table: ttab[t] = kd*HW + kh*W + kw  (unsigned: keeps address math 32-bit)
__global__ void ttab_kernel(unsigned* __restrict__ ttab, int KV, int HW, int W, int KH, int KW) {
  int t = blockIdx.x * blockDim.x + threadIdx.x;
  if (t >= KV) return;
  int kd = t / (KH * KW);
  int r  = t - kd * (KH * KW);
  int kh = r / KW;
  int kw = r - kh * KW;
  ttab[t] = (unsigned)(kd * HW + kh * W + kw);
}

// ---------------- conv1: [64,1,92^3] f32 -> [64,44^3,16] f16 (channels-last) -------------
// stride 2, kernel 6^3. Paired-kd K-enumeration => every tap offset is a compile-time
// immediate; per-lane base = x + spatial_base + lh*HW. B loads are float2 with static
// IOFFSETs, A loads are 2x b128 with static IOFFSETs. Fully unrolled (7 chunks).
__global__ void conv1_wmma_kernel(const float* __restrict__ x,
                                  const _Float16* __restrict__ wgt,
                                  const float* __restrict__ bias,
                                  _Float16* __restrict__ out_h, int nst) {
  constexpr int W = 92, HW = 92 * 92, DHW = 92 * HW;
  constexpr int OW = 44, OHW = 44 * 44, S = 44 * OHW;

  const int wid = (blockIdx.x * blockDim.x + threadIdx.x) >> 5;
  if (wid >= 64 * nst) return;  // wave-uniform
  const int st = wid % nst, n = wid / nst;
  const int lane = threadIdx.x & 31, lh = lane >> 4, lm = lane & 15;

  const int sp = st * 16 + lm;  // S is a multiple of 16 => always in range
  int od = sp / OHW;
  int r  = sp - od * OHW;
  int oh = r / OW;
  int ow = r - oh * OW;
  const float* xp = x + ((unsigned)(n * DHW + 2 * od * HW + 2 * oh * W + 2 * ow) +
                         (unsigned)(lh * HW));
  const _Float16* ap = wgt + (unsigned)(lm * 224 + lh * 16);

  v8f acc = {};
#pragma unroll
  for (int kc = 0; kc < 224; kc += 32) {
    const int P = kc / 32;
    v16h a, b;
    v8h q0 = *(const v8h*)(ap + kc);
    v8h q1 = *(const v8h*)(ap + kc + 8);
#pragma unroll
    for (int i = 0; i < 8; ++i) { a[i] = q0[i]; a[i + 8] = q1[i]; }
#pragma unroll
    for (int i = 0; i < 16; i += 2) {
      const int J = P * 16 + i;          // compile-time
      if (J < 108) {
        const int aa  = J / 36, rem = J % 36;
        const int off = 2 * aa * HW + (rem / 6) * W + (rem % 6);  // compile-time immediate
        float2 f = *(const float2*)(xp + off);                    // kw even => contiguous
        b[i]     = (_Float16)f.x;
        b[i + 1] = (_Float16)f.y;
      } else {                            // pad taps (J in [108,112)): compile-time zeros
        b[i]     = (_Float16)0.f;
        b[i + 1] = (_Float16)0.f;
      }
    }
    acc = __builtin_amdgcn_wmma_f32_16x16x32_f16(false, a, false, b, (short)0, acc,
                                                 false, false);
  }

  v8h o;
#pragma unroll
  for (int rr = 0; rr < 8; ++rr)
    o[rr] = (_Float16)leaky01(acc[rr] + bias[lh * 8 + rr]);
  *(v8h*)(out_h + (unsigned)((n * S + sp) * 16 + lh * 8)) = o;  // packed b128 store
}

// ---------------- channels-last conv via WMMA, register-blocked over Cout tiles ----------
// in: [N][D][H][W][CIN] f16; k enumerated as (tap, c); every 16-run of k is contiguous.
template <int NMT, int CIN>
__global__ void convcl_wmma_kernel(
    const _Float16* __restrict__ in, const _Float16* __restrict__ wgt,
    const float* __restrict__ bias, const unsigned* __restrict__ ttab,
    _Float16* __restrict__ out_h, float* __restrict__ out_f,
    int N, int D, int H, int W, int OD, int OH, int OW,
    int stride, int KV, int Kpad, int nst) {
  __shared__ unsigned ttab_s[128];
  for (int i = threadIdx.x; i < KV; i += blockDim.x) ttab_s[i] = ttab[i];
  __syncthreads();

  const int S = OD * OH * OW;
  const int wid = (blockIdx.x * blockDim.x + threadIdx.x) >> 5;
  if (wid >= N * nst) return;  // wave-uniform
  const int st = wid % nst, n = wid / nst;
  const int lane = threadIdx.x & 31, lh = lane >> 4, lm = lane & 15;
  const int Cout = NMT * 16;

  const int sp  = st * 16 + lm;
  const int spc = (sp < S) ? sp : 0;
  int od = spc / (OH * OW);
  int r  = spc - od * (OH * OW);
  int oh = r / OW;
  int ow = r - oh * OW;
  const int HW = H * W;
  const unsigned base2 = (unsigned)(n * D * HW + stride * od * HW + stride * oh * W + stride * ow);

  const int khb = lh * 16, kha = lh * 16;

  v8f acc[NMT] = {};
  for (int kc = 0; kc < Kpad; kc += 32) {
    // B fragment: one contiguous 16-half run per lane (single tap, CIN multiple of 16)
    int k  = kc + khb;
    int t  = k / CIN;            // compile-time divisor
    unsigned c0 = (unsigned)(k - t * CIN);
    unsigned toff = (t < KV) ? ttab_s[t] : 0u;  // pad region: clamped in-tensor, A is zero there
    const _Float16* bp = in + ((base2 + toff) * (unsigned)CIN + c0);
    v16h b;
    v8h p0 = *(const v8h*)bp;
    v8h p1 = *(const v8h*)(bp + 8);
#pragma unroll
    for (int i = 0; i < 8; ++i) { b[i] = p0[i]; b[i + 8] = p1[i]; }

    // prefetch next chunk's B run
    int tn = (k + 32) / CIN;
    if (tn >= KV) tn = KV - 1;
    __builtin_prefetch(in + (base2 + ttab_s[tn]) * (unsigned)CIN, 0, 1);

#pragma unroll
    for (int m = 0; m < NMT; ++m) {  // B reused across NMT WMMAs
      const _Float16* ap = wgt + (unsigned)((m * 16 + lm) * Kpad + kc + kha);
      v8h q0 = *(const v8h*)ap;
      v8h q1 = *(const v8h*)(ap + 8);
      v16h a;
#pragma unroll
      for (int i = 0; i < 8; ++i) { a[i] = q0[i]; a[i + 8] = q1[i]; }
      acc[m] = __builtin_amdgcn_wmma_f32_16x16x32_f16(false, a, false, b, (short)0,
                                                      acc[m], false, false);
    }
  }

  if (sp < S) {
#pragma unroll
    for (int m = 0; m < NMT; ++m) {
      v8h o;
      float vf[8];
#pragma unroll
      for (int rr = 0; rr < 8; ++rr) {
        float v = leaky01(acc[m][rr] + bias[m * 16 + lh * 8 + rr]);
        vf[rr] = v;
        o[rr] = (_Float16)v;
      }
      unsigned ob = (unsigned)((n * S + sp) * Cout + m * 16 + lh * 8);
      *(v8h*)(out_h + ob) = o;
      if (out_f) {
        float4 f0 = {vf[0], vf[1], vf[2], vf[3]};
        float4 f1 = {vf[4], vf[5], vf[6], vf[7]};
        *(float4*)(out_f + ob)     = f0;
        *(float4*)(out_f + ob + 4) = f1;
      }
    }
  }
}

// ---------------- GEMM: M = feature[64,512] @ T[512,1024] (B pre-transposed) ------------
__global__ void gemm_wmma_kernel(const _Float16* __restrict__ A,   // [64][512]
                                 const _Float16* __restrict__ Bt,  // [1024][512]
                                 float* __restrict__ C) {          // [64][1024]
  const int wid = (blockIdx.x * blockDim.x + threadIdx.x) >> 5;
  if (wid >= 64) return;  // one N-tile per wave, all 4 M-tiles register-blocked
  const int nt = wid;
  const int lane = threadIdx.x & 31, lh = lane >> 4, lm = lane & 15;
  const int khb = lh * 16, kha = lh * 16;

  const _Float16* brow = Bt + (unsigned)((nt * 16 + lm) * 512 + khb);
  v8f acc[4] = {};
  for (int kc = 0; kc < 512; kc += 32) {
    v16h b;
    v8h p0 = *(const v8h*)(brow + kc);
    v8h p1 = *(const v8h*)(brow + kc + 8);
#pragma unroll
    for (int i = 0; i < 8; ++i) { b[i] = p0[i]; b[i + 8] = p1[i]; }
#pragma unroll
    for (int m = 0; m < 4; ++m) {
      const _Float16* ap = A + (unsigned)((m * 16 + lm) * 512 + kc + kha);
      v8h q0 = *(const v8h*)ap;
      v8h q1 = *(const v8h*)(ap + 8);
      v16h a;
#pragma unroll
      for (int i = 0; i < 8; ++i) { a[i] = q0[i]; a[i + 8] = q1[i]; }
      acc[m] = __builtin_amdgcn_wmma_f32_16x16x32_f16(false, a, false, b, (short)0,
                                                      acc[m], false, false);
    }
  }
#pragma unroll
  for (int m = 0; m < 4; ++m)
#pragma unroll
    for (int rr = 0; rr < 8; ++rr)
      C[(unsigned)((m * 16 + lh * 8 + rr) * 1024 + nt * 16 + lm)] = acc[m][rr];
}

// ---------------- feature repack: [n][sp(8)][c(64)] f16 -> [n][c*8+sp] f16 --------------
__global__ void repack_feat_kernel(const _Float16* __restrict__ h4, _Float16* __restrict__ f) {
  int idx = blockIdx.x * blockDim.x + threadIdx.x;
  if (idx >= 64 * 512) return;
  int n = idx >> 9, q = idx & 511, c = q >> 3, sp = q & 7;
  f[idx] = h4[(n * 8 + sp) * 64 + c];
}

// ---------------- minibatch discrimination ----------------------------------------------
__global__ void minibatch_kernel(const float* __restrict__ Mm, float* __restrict__ outT, int N) {
  const int i = blockIdx.x, c = threadIdx.x;
  float mi[16];
#pragma unroll
  for (int k = 0; k < 16; ++k) mi[k] = Mm[i * 1024 + c * 16 + k];
  float acc = 0.f;
  for (int j = 0; j < N; ++j) {
    float d = 0.f;
#pragma unroll
    for (int k = 0; k < 16; ++k) d += fabsf(mi[k] - Mm[j * 1024 + c * 16 + k]);
    acc += expf(d);
  }
  outT[i * 64 + c] = acc;
}

// ---------------- head: sigmoid(concat(feature, out_T) @ Wm + bm) ------------------------
// featf is channels-last [n][sp(8)][c(64)] f32
__global__ void head_kernel(const float* __restrict__ featf, const float* __restrict__ outT,
                            const float* __restrict__ Wm, const float* __restrict__ bm,
                            float* __restrict__ out, int N) {
  int n = threadIdx.x;
  if (n >= N) return;
  float z = bm[0];
  for (int c = 0; c < 64; ++c)
    for (int sp = 0; sp < 8; ++sp)
      z += featf[(n * 8 + sp) * 64 + c] * Wm[c * 8 + sp];
  for (int c = 0; c < 64; ++c) z += outT[n * 64 + c] * Wm[512 + c];
  out[n] = 1.f / (1.f + expf(-z));
}

// ---------------- launcher ----------------
extern "C" void kernel_launch(void* const* d_in, const int* in_sizes, int n_in,
                              void* d_out, int out_size, void* d_ws, size_t ws_size,
                              hipStream_t stream) {
  (void)in_sizes; (void)n_in; (void)out_size; (void)ws_size;
  const float* x  = (const float*)d_in[0];
  const float* w1 = (const float*)d_in[1];
  const float* b1 = (const float*)d_in[2];
  const float* w2 = (const float*)d_in[3];
  const float* b2 = (const float*)d_in[4];
  const float* w3 = (const float*)d_in[5];
  const float* b3 = (const float*)d_in[6];
  const float* w4 = (const float*)d_in[7];
  const float* b4 = (const float*)d_in[8];
  const float* T  = (const float*)d_in[9];
  const float* Wm = (const float*)d_in[10];
  const float* bm = (const float*)d_in[11];
  float* out = (float*)d_out;

  char* ws = (char*)d_ws;
  size_t off = 0;
  auto alloc = [&](size_t bytes) -> void* {
    void* p = ws + off;
    off = (off + bytes + 255) & ~(size_t)255;
    return p;
  };

  const int N = 64;
  // channels-last activations (f16)
  _Float16* h1    = (_Float16*)alloc((size_t)N * 85184 * 16 * 2);  // [n][44^3][16]
  _Float16* h2    = (_Float16*)alloc((size_t)N * 2744 * 32 * 2);   // [n][14^3][32]
  _Float16* h3    = (_Float16*)alloc((size_t)N * 64 * 48 * 2);     // [n][4^3][48]
  _Float16* h4cl  = (_Float16*)alloc((size_t)N * 8 * 64 * 2);      // [n][8][64]
  float*    featf = (float*)   alloc((size_t)N * 8 * 64 * 4);      // [n][8][64] f32
  _Float16* feath = (_Float16*)alloc((size_t)N * 512 * 2);         // [n][512] std order
  // f16 weights / T
  _Float16* w1h = (_Float16*)alloc((size_t)16 * 224 * 2);
  _Float16* w2h = (_Float16*)alloc((size_t)32 * 2016 * 2);
  _Float16* w3h = (_Float16*)alloc((size_t)48 * 4000 * 2);
  _Float16* w4h = (_Float16*)alloc((size_t)64 * 1312 * 2);
  _Float16* Tt  = (_Float16*)alloc((size_t)1024 * 512 * 2);
  // tap tables (conv2-4)
  unsigned* tt2 = (unsigned*)alloc(128 * 4);
  unsigned* tt3 = (unsigned*)alloc(128 * 4);
  unsigned* tt4 = (unsigned*)alloc(128 * 4);
  // tail
  float* Mmat = (float*)alloc((size_t)N * 1024 * 4);
  float* outT = (float*)alloc((size_t)N * 64 * 4);

  // --- prep ---
  wpad1_kernel<<<(16 * 224 + 255) / 256, 256, 0, stream>>>(w1, w1h);
  wpad_cl_kernel<<<(32 * 2016 + 255) / 256, 256, 0, stream>>>(w2, w2h, 32, 16, 125, 2016);
  wpad_cl_kernel<<<(48 * 4000 + 255) / 256, 256, 0, stream>>>(w3, w3h, 48, 32, 125, 4000);
  wpad_cl_kernel<<<(64 * 1312 + 255) / 256, 256, 0, stream>>>(w4, w4h, 64, 48, 27, 1312);
  tpose_f16_kernel<<<(1024 * 512 + 255) / 256, 256, 0, stream>>>(T, Tt);
  ttab_kernel<<<1, 128, 0, stream>>>(tt2, 125, 44 * 44, 44, 5, 5);
  ttab_kernel<<<1, 128, 0, stream>>>(tt3, 125, 14 * 14, 14, 5, 5);
  ttab_kernel<<<1, 128, 0, stream>>>(tt4, 27, 4 * 4, 4, 3, 3);

  // --- conv1: f32 x -> h1 ---
  {
    int nst = 85184 / 16;                 // 5324
    int waves = N * nst;                  // 340736
    conv1_wmma_kernel<<<(waves + 3) / 4, 128, 0, stream>>>(x, w1h, b1, h1, nst);
  }
  // --- conv2: h1 -> h2 ---
  {
    int nst = (2744 + 15) / 16;           // 172
    int waves = N * nst;                  // 11008
    convcl_wmma_kernel<2, 16><<<(waves + 3) / 4, 128, 0, stream>>>(
        h1, w2h, b2, tt2, h2, (float*)nullptr,
        N, 44, 44, 44, 14, 14, 14, 3, 125, 2016, nst);
  }
  // --- conv3: h2 -> h3 ---
  {
    int nst = 4;
    int waves = N * nst;                  // 256
    convcl_wmma_kernel<3, 32><<<(waves + 3) / 4, 128, 0, stream>>>(
        h2, w3h, b3, tt3, h3, (float*)nullptr,
        N, 14, 14, 14, 4, 4, 4, 3, 125, 4000, nst);
  }
  // --- conv4: h3 -> h4cl (+f32 feature, channels-last) ---
  {
    int nst = 1;                          // S=8 partial tile
    int waves = N * nst;                  // 64
    convcl_wmma_kernel<4, 48><<<(waves + 3) / 4, 128, 0, stream>>>(
        h3, w4h, b4, tt4, h4cl, featf,
        N, 4, 4, 4, 2, 2, 2, 1, 27, 1312, nst);
  }

  // --- feature repack + M = feature @ T ---
  repack_feat_kernel<<<(64 * 512 + 255) / 256, 256, 0, stream>>>(h4cl, feath);
  gemm_wmma_kernel<<<16, 128, 0, stream>>>(feath, Tt, Mmat);

  // --- minibatch discrimination + head ---
  minibatch_kernel<<<64, 64, 0, stream>>>(Mmat, outT, N);
  head_kernel<<<1, 64, 0, stream>>>(featf, outT, Wm, bm, out, N);
}
